// AttentionEssentialReinforce_51238959841470
// MI455X (gfx1250) — compile-verified, matrix operand order checked by reading, not run on hardware
//
#include <hip/hip_runtime.h>
#include <stdint.h>

// CDNA5 / gfx1250 — wave32, WMMA, 320KB LDS per WGP.
typedef __attribute__((ext_vector_type(16))) _Float16 v16h;
typedef __attribute__((ext_vector_type(8)))  float    v8f;

#define ROWLEN   4096
#define ROWLEN2  8192
#define TPB      256
#define MASKIDF  103.0f

// Order-preserving float32 -> uint32 mapping (ascending).
__device__ __forceinline__ unsigned ordkey(float f) {
  unsigned u = __float_as_uint(f);
  return (u & 0x80000000u) ? ~u : (u | 0x80000000u);
}

__global__ __launch_bounds__(TPB)
void mlm_gumbel_topk_kernel(const int* __restrict__ ids,
                            const float* __restrict__ wmask,
                            const float* __restrict__ uin,
                            float* __restrict__ out,
                            int nrows) {
  __shared__ unsigned long long keys[ROWLEN];   // 32 KB: full row of sortable keys
  __shared__ int      hist[256];
  __shared__ float    s_cnt;
  __shared__ unsigned s_digit;
  __shared__ int      s_kk;

  const int row = blockIdx.x;
  if (row >= nrows) return;
  const int t = threadIdx.x;

  const int*   idp = ids   + (size_t)row * ROWLEN;
  const float* wp  = wmask + (size_t)row * ROWLEN2;   // first half = w
  const float* up  = uin   + (size_t)row * ROWLEN;

  if (t == 0) s_cnt = 0.0f;
  __syncthreads();

  // ---- Phase 1: coalesced b128 loads, compute Gumbel keys into LDS ----
  int idr[16];
  unsigned nzbits = 0;
  int mycnt = 0;
#pragma unroll
  for (int j = 0; j < 4; ++j) {
    const int i0 = j * (TPB * 4) + t * 4;            // fully coalesced float4/int4
    float4 wv = *(const float4*)(wp + i0);
    float4 uv = *(const float4*)(up + i0);
    int4   iv = *(const int4*)(idp + i0);
    const float wa[4] = {wv.x, wv.y, wv.z, wv.w};
    const float ua[4] = {uv.x, uv.y, uv.z, uv.w};
    const int   ia[4] = {iv.x, iv.y, iv.z, iv.w};
#pragma unroll
    for (int e = 0; e < 4; ++e) {
      const int i = i0 + e;
      const bool nz = wa[e] > 0.0f;
      const float g  = -logf(-logf(ua[e]));          // Gumbel noise
      const float sc = logf(fmaxf(wa[e], 1e-30f)) + g;
      const unsigned k32 = nz ? ordkey(sc) : 0u;
      // 64-bit key: score in high word, inverted index low -> all keys distinct,
      // ties broken by smaller index (matches stable argsort rank).
      keys[i] = ((unsigned long long)k32 << 32) | (unsigned)(0xFFFFFFFFu - (unsigned)i);
      idr[j * 4 + e] = ia[e];
      nzbits |= (nz ? 1u : 0u) << (j * 4 + e);
      mycnt  += nz ? 1 : 0;
    }
  }
  __syncthreads();

  // ---- Phase 2: wave32 count reduction on the matrix pipe ----
  // D = A(16x32 f16) x ones(32x16) + 0. Every column of D sums all 512 A
  // elements across the two row-halves, so lanes 0 and 16 (rows 0-7 / 8-15 of
  // column 0) together hold the exact wave total. One v_wmma per wave.
  {
    v16h a;
#pragma unroll
    for (int i = 0; i < 16; ++i) a[i] = (_Float16)0.0f;
    a[0] = (_Float16)(float)mycnt;
    v16h b;
#pragma unroll
    for (int i = 0; i < 16; ++i) b[i] = (_Float16)1.0f;
    v8f c = {};
    c = __builtin_amdgcn_wmma_f32_16x16x32_f16(false, a, false, b,
                                               (short)0, c, false, false);
    const float s = c[0] + c[1] + c[2] + c[3] + c[4] + c[5] + c[6] + c[7];
    if ((t & 15) == 0) atomicAdd(&s_cnt, s);       // lanes 0 and 16 of each wave
  }
  __syncthreads();

  const int cnt = (int)(s_cnt + 0.5f);
  const int k   = (int)floorf(0.15f * (float)cnt); // MU_P, f32 math like reference

  // ---- Phase 3: exact k-th largest via MSB-first LDS radix select ----
  unsigned long long kth = ~0ULL;                   // k==0 -> nothing selected
  if (k > 0) {
    if (t == 0) s_kk = k;
    unsigned long long wantHi = 0;                  // known high digits so far
    for (int p = 7; p >= 0; --p) {
      hist[t] = 0;
      __syncthreads();
      const int sh = 8 * p;
#pragma unroll
      for (int j = 0; j < 4; ++j) {
        const int i0 = j * (TPB * 4) + t * 4;
#pragma unroll
        for (int e = 0; e < 4; ++e) {
          const unsigned long long key = keys[i0 + e];
          const bool cand = (p == 7) || ((key >> (sh + 8)) == wantHi);
          if (cand) atomicAdd(&hist[(int)((key >> sh) & 255u)], 1);
        }
      }
      __syncthreads();
      if (t == 0) {
        int kk = s_kk, c = 0, d = 255;
        for (; d > 0; --d) {                        // scan digits high->low
          const int h = hist[d];
          if (c + h >= kk) break;
          c += h;
        }
        s_digit = (unsigned)d;
        s_kk    = kk - c;                           // rank within chosen bucket
      }
      __syncthreads();
      wantHi = (wantHi << 8) | s_digit;
    }
    kth = wantHi;                                   // exact k-th largest key
  }

  // ---- Phase 4: coalesced b128 stores of (ids, mask, -mask) ----
  const size_t Ntot = (size_t)nrows * ROWLEN;
  float* o_ids = out;
  float* o_m   = out + Ntot;
  float* o_n   = out + 2 * Ntot;
  const size_t rb = (size_t)row * ROWLEN;
#pragma unroll
  for (int j = 0; j < 4; ++j) {
    const int i0 = j * (TPB * 4) + t * 4;
    float r0[4], r1[4], r2[4];
#pragma unroll
    for (int e = 0; e < 4; ++e) {
      const int li = j * 4 + e;
      const bool nz  = (nzbits >> li) & 1u;
      const bool sel = nz && (keys[i0 + e] >= kth); // exactly k selected
      r0[e] = sel ? MASKIDF : (float)idr[li];
      r1[e] = sel ? 1.0f : 0.0f;
      r2[e] = -r1[e];
    }
    *(float4*)(o_ids + rb + i0) = make_float4(r0[0], r0[1], r0[2], r0[3]);
    *(float4*)(o_m   + rb + i0) = make_float4(r1[0], r1[1], r1[2], r1[3]);
    *(float4*)(o_n   + rb + i0) = make_float4(r2[0], r2[1], r2[2], r2[3]);
  }
}

extern "C" void kernel_launch(void* const* d_in, const int* in_sizes, int n_in,
                              void* d_out, int out_size, void* d_ws, size_t ws_size,
                              hipStream_t stream) {
  const int*   ids = (const int*)d_in[0];    // input_ids  (B,J,L)  int32
  const float* wm  = (const float*)d_in[1];  // my_attention_mask (B,J,2L) f32
  const float* uu  = (const float*)d_in[2];  // u (B,J,L) f32
  float* out = (float*)d_out;                // [ids | mask | -mask], 3*B*J*L f32

  const int nrows = in_sizes[0] / ROWLEN;    // B*J = 512
  mlm_gumbel_topk_kernel<<<nrows, TPB, 0, stream>>>(ids, wm, uu, out, nrows);
}
